// MuskingumCungeRouter_71829033058762
// MI455X (gfx1250) — matrix operand order repaired and checked by reading, not compiled, not run on hardware
//
#include <hip/hip_runtime.h>
#include <hip/hip_bf16.h>
#include <math.h>

typedef __attribute__((ext_vector_type(16))) _Float16 v16h;
typedef __attribute__((ext_vector_type(8)))  float    v8f;

#define N_REACH   4096
#define N_STEPS   96
#define DT_S      3600.0f
#define NBLK      256          // 4096 / 16 row-blocks
#define NKT       128          // 4096 / 32 K-tiles per full row
#define TILE_H    512          // halfs per 16x32 A tile (32 lanes * 16 halfs)
#define NWAVE     16           // 512-thread solver block = 16 wave32
#define WS_A_BYTES ((size_t)NBLK * NKT * TILE_H * 2)   // 33,554,432 B of f16 A tiles

// ---------------------------------------------------------------------------
// Prep: convert int32 0/1 mask -> f16 tiles in WMMA A-operand layout.
// A (16x32 f16) layout used here (contiguous k order): lane L holds row M=L&15; half h maps to
//   K = h + (L>=16 ? 16 : 0)
// Tile (bm,bk) covers rows [16bm,16bm+16), cols [32bk,32bk+32).
// ---------------------------------------------------------------------------
__global__ __launch_bounds__(256) void mc_prep(const int* __restrict__ mask,
                                               _Float16* __restrict__ wsA) {
    unsigned idx = blockIdx.x * 256u + threadIdx.x;           // < 16,777,216
    unsigned t    = idx >> 9;            // tile id
    unsigned r    = idx & 511u;          // element in tile
    unsigned ln   = r >> 4;              // lane 0..31
    unsigned h    = r & 15u;             // half 0..15
    unsigned bm   = t >> 7;              // row block
    unsigned bk   = t & 127u;            // k tile
    unsigned row  = (bm << 4) + (ln & 15u);
    unsigned K    = h + ((ln >> 4) << 4);
    unsigned col  = (bk << 5) + K;
    int m = mask[(size_t)row * N_REACH + col];
    wsA[idx] = m ? (_Float16)1.0f : (_Float16)0.0f;
}

// Per-block flag: does the strictly-lower 16x16 diagonal block contain any link?
__global__ void mc_diag(const int* __restrict__ mask, int* __restrict__ wsDiag) {
    int bm = threadIdx.x;                 // 256 threads, one per block
    int base = bm << 4;
    int flag = 0;
    for (int r = 1; r < 16; ++r)
        for (int s = 0; s < r; ++s)
            flag |= (mask[(size_t)(base + r) * N_REACH + base + s] != 0);
    wsDiag[bm] = flag;
}

__device__ __forceinline__ float safe_pow_f(float b, float e) {
    return __expf(e * __logf(fmaxf(b, 1e-6f)));
}

// ---------------------------------------------------------------------------
// Persistent single-workgroup solver: 96 timesteps x 256 block-sequential
// triangular sweep. GEMV panels run on v_wmma_f32_16x16x32_f16 with an
// unroll-2 double-buffered tile loop and dual accumulators, so each WMMA
// executes with the other buffer's loads in flight (partial waits) and the
// WMMA->WMMA accumulator RAW hazard is spaced out.
// ---------------------------------------------------------------------------
__global__ __launch_bounds__(512) void mc_solver(
    const float* __restrict__ lateral,   // [96*4096]
    const float* __restrict__ initQ,
    const float* __restrict__ Lr, const float* __restrict__ Sr,
    const float* __restrict__ mnr, const float* __restrict__ wcr,
    const float* __restrict__ wer, const float* __restrict__ dcr,
    const float* __restrict__ der,
    const int*   __restrict__ mask,      // original int mask (diag blocks)
    const int*   __restrict__ is_outlet,
    const _Float16* __restrict__ wsA,    // f16 A tiles
    const int*   __restrict__ wsDiag,
    float* __restrict__ gQprev,          // [4096] state: Q_prev
    float* __restrict__ gIprev,          // [4096] state: I_prev
    float* __restrict__ out)             // [96]
{
    __shared__ __align__(32) _Float16 sQ16[N_REACH];   // 8 KB  f16 Q (B operand)
    __shared__ __align__(32) _Float16 sZero[16];       // 32 B  zero pad for B
    __shared__ float sQ32[N_REACH];                    // 16 KB current-step Q
    __shared__ float sC0 [N_REACH];                    // 16 KB normalized C0
    __shared__ float sB  [N_REACH];                    // 16 KB C1*Iprev + C2*Qprev
    __shared__ float sPart[NWAVE * 16];                // 1 KB  per-wave partials
    __shared__ float sRed[512];                        // 2 KB  outlet reduce

    const int tid  = threadIdx.x;
    const int lane = tid & 31;
    const int wS   = __builtin_amdgcn_readfirstlane(tid >> 5);  // scalar wave id

    // Per-lane B-operand source rule (constant across all tiles):
    //   lane 0  -> sQ16 + col0      (K = 0..15,  N = 0)
    //   lane 16 -> sQ16 + col0 + 16 (K = 16..31, N = 0)
    //   others  -> sZero            (columns 1..15 are don't-care zeros)
    const bool useZero = (lane != 0) && (lane != 16);
    const _Float16* bbase = (lane == 16) ? (sQ16 + 16) : sQ16;

    // ---- initial state: (Q, Q_prev, I_prev) = (initQ, initQ, 0) ----
    if (tid < 16) sZero[tid] = (_Float16)0.0f;
    for (int k = 0; k < 8; ++k) {
        int i = (tid << 3) + k;
        float q = initQ[i];
        sQ32[i]  = q;
        gQprev[i] = q;
        gIprev[i] = 0.0f;
    }
    __syncthreads();

    for (int t = 0; t < N_STEPS; ++t) {
        // ---- phase 1: Muskingum-Cunge coefficients (elementwise) ----
        for (int k = 0; k < 8; ++k) {
            int i = (tid << 3) + k;
            float Qst  = sQ32[i];                 // state.Q
            float Qref = fmaxf(Qst, 0.1f);
            float Qs   = fmaxf(Qref, 0.01f);
            float W  = wcr[i] * safe_pow_f(Qs, wer[i]);
            float Dd = dcr[i] * safe_pow_f(Qs, der[i]);
            float A  = W * Dd;
            float R  = A / (W + 2.0f * Dd);
            float Ss = fmaxf(Sr[i], 1e-6f);
            float V  = (1.0f / fmaxf(mnr[i], 1e-3f)) * safe_pow_f(R, 0.66666667f) * sqrtf(Ss);
            float c  = fmaxf(1.66666667f * V, 0.01f);
            float K  = fmaxf(Lr[i] / c, DT_S * 0.1f);
            float X  = 0.5f * (1.0f - Qref / (W * c * Ss * Lr[i] + 1e-6f));
            X = fminf(fmaxf(X, 0.0f), 0.5f);
            float den = 2.0f * K * (1.0f - X) + DT_S;
            float C0 = fmaxf((DT_S - 2.0f * K * X) / den, 0.0f);
            float C1 = fmaxf((DT_S + 2.0f * K * X) / den, 0.0f);
            float C2 = fmaxf((2.0f * K * (1.0f - X) - DT_S) / den, 0.0f);
            float tot = C0 + C1 + C2;
            sC0[i] = C0 / tot;
            sB[i]  = (C1 / tot) * gIprev[i] + (C2 / tot) * gQprev[i];
            gQprev[i] = Qst;                      // becomes Q_prev for step t+1
            sQ16[i] = (_Float16)0.0f;             // Q_acc starts at zero (ref)
        }
        __syncthreads();

        const float* lat_t = lateral + t * N_REACH;

        // ---- phase 2: blocked triangular sweep ----
        for (int bm = 0; bm < NBLK; ++bm) {
            // -- 2a: inter-block GEMV via WMMA, K-tiles split across waves --
            v8f acc0 = {}, acc1 = {};
            const int nk = (bm >> 1) + 1;         // tiles covering cols < 16bm+16
            const size_t step = (size_t)NWAVE << 9;
            const _Float16* at =
                wsA + (((size_t)(bm << 7) + wS) << 9) + (lane << 4);
            int bk  = wS;
            int cnt = (wS < nk) ? ((nk - 1 - wS) / NWAVE + 1) : 0;
            if (cnt > 0) {
                v16h a0 = *(const v16h*)at;
                v16h b0 = *(const v16h*)(useZero ? sZero : (bbase + (bk << 5)));
                while (cnt >= 2) {
                    __builtin_prefetch(at + 2 * step, 0, 3);
                    v16h a1 = *(const v16h*)(at + step);
                    v16h b1 = *(const v16h*)(useZero ? sZero
                                                     : (bbase + ((bk + NWAVE) << 5)));
                    acc0 = __builtin_amdgcn_wmma_f32_16x16x32_f16(
                            false, a0, false, b0, (short)0, acc0, false, false);
                    at += 2 * step; bk += 2 * NWAVE; cnt -= 2;
                    if (cnt > 0) {
                        a0 = *(const v16h*)at;
                        b0 = *(const v16h*)(useZero ? sZero : (bbase + (bk << 5)));
                    }
                    acc1 = __builtin_amdgcn_wmma_f32_16x16x32_f16(
                            false, a1, false, b1, (short)0, acc1, false, false);
                }
                if (cnt == 1) {
                    acc0 = __builtin_amdgcn_wmma_f32_16x16x32_f16(
                            false, a0, false, b0, (short)0, acc0, false, false);
                }
            }
            #pragma unroll
            for (int r = 0; r < 8; ++r) acc0[r] += acc1[r];

            // D column N=0 lives in lane 0 (M=0..7) and lane 16 (M=8..15)
            if (lane == 0) {
                #pragma unroll
                for (int r = 0; r < 8; ++r) sPart[wS * 16 + r] = acc0[r];
            } else if (lane == 16) {
                #pragma unroll
                for (int r = 0; r < 8; ++r) sPart[wS * 16 + 8 + r] = acc0[r];
            }
            __syncthreads();

            // -- 2b: intra-block finalize (16 lanes of wave 0) --
            if (tid < 16) {
                const int i = (bm << 4) + tid;
                float p = 0.0f;
                #pragma unroll
                for (int ww = 0; ww < NWAVE; ++ww) p += sPart[ww * 16 + tid];
                const float c0 = sC0[i], bv = sB[i], lat = lat_t[i];
                float myQ = 0.0f, myI = 0.0f;
                if (wsDiag[bm]) {
                    // serial 16-step chain (rare: mask density ~4/4096)
                    unsigned mb = 0;
                    const int* mrow = mask + (size_t)i * N_REACH + (bm << 4);
                    #pragma unroll
                    for (int s = 0; s < 16; ++s)
                        mb |= (mrow[s] != 0 ? 1u : 0u) << s;
                    for (int s = 0; s < 16; ++s) {
                        float Ic = p + lat;
                        float qo = fmaxf(fmaf(c0, Ic, bv), 0.0f);
                        float qb = __int_as_float(
                            __builtin_amdgcn_ds_bpermute(s << 2, __float_as_int(qo)));
                        if (tid == s) { myQ = qb; myI = Ic; }
                        if (tid > s && ((mb >> s) & 1u)) p += qb;
                    }
                } else {
                    myI = p + lat;
                    myQ = fmaxf(fmaf(c0, myI, bv), 0.0f);
                }
                sQ32[i] = myQ;
                sQ16[i] = (_Float16)myQ;
                gIprev[i] = myI;                  // I_prev for step t+1
            }
            __syncthreads();
        }

        // ---- phase 3: outlet (masked sum over is_outlet) ----
        float ls = 0.0f;
        for (int k = 0; k < 8; ++k) {
            int i = (tid << 3) + k;
            if (is_outlet[i]) ls += sQ32[i];
        }
        sRed[tid] = ls;
        __syncthreads();
        if (tid == 0) {
            float ssum = 0.0f;
            for (int j = 0; j < 512; ++j) ssum += sRed[j];
            out[t] = ssum;
        }
        __syncthreads();
    }
}

extern "C" void kernel_launch(void* const* d_in, const int* in_sizes, int n_in,
                              void* d_out, int out_size, void* d_ws, size_t ws_size,
                              hipStream_t stream) {
    const float* lateral = (const float*)d_in[0];   // [96,4096]
    const float* initQ   = (const float*)d_in[1];
    const float* lengths = (const float*)d_in[2];
    const float* slopes  = (const float*)d_in[3];
    const float* mann    = (const float*)d_in[4];
    const float* wcoef   = (const float*)d_in[5];
    const float* wexp    = (const float*)d_in[6];
    const float* dcoef   = (const float*)d_in[7];
    const float* dexp    = (const float*)d_in[8];
    const int*   mask    = (const int*)d_in[9];     // [4096,4096] 0/1
    const int*   outlet  = (const int*)d_in[10];    // [4096] 0/1

    char* ws = (char*)d_ws;
    _Float16* wsA    = (_Float16*)ws;                         // 32 MB A tiles
    int*      wsDiag = (int*)(ws + WS_A_BYTES);               // 1 KB flags
    float*    gQprev = (float*)(ws + WS_A_BYTES + 4096);      // 16 KB
    float*    gIprev = (float*)(ws + WS_A_BYTES + 4096 + 16384);
    float*    out    = (float*)d_out;

    // 1) mask -> f16 WMMA A-layout tiles (L2-resident thereafter)
    mc_prep<<<65536, 256, 0, stream>>>(mask, wsA);
    // 2) per-block diagonal-link flags
    mc_diag<<<1, 256, 0, stream>>>(mask, wsDiag);
    // 3) persistent single-workgroup router
    mc_solver<<<1, 512, 0, stream>>>(lateral, initQ, lengths, slopes, mann,
                                     wcoef, wexp, dcoef, dexp, mask, outlet,
                                     wsA, wsDiag, gQprev, gIprev, out);
}
